// Discriminator_89120571392391
// MI455X (gfx1250) — matrix operand order repaired and checked by reading, not compiled, not run on hardware
//
#include <hip/hip_runtime.h>
#include <hip/hip_bf16.h>

// ---------------------------------------------------------------------------
// 2-layer GRU scan (B=256, H=512, T=2048) as one persistent CDNA5 kernel.
// bf16 WMMA (v_wmma_f32_16x16x32_bf16) for the recurrent GEMMs, weights
// resident in LDS (48 KB strip per WGP), device-wide barriers between the
// GEMM and pointwise phases of each timestep.
// ---------------------------------------------------------------------------

#define BSZ     256
#define HID     512
#define G3      1536          // 3*HID
#define TSTEPS  2048
#define NWG     96            // persistent workgroups
#define NSTRIP  16            // gate columns per workgroup (G3 / NWG)
#define NTHREADS 256          // 8 wave32 per workgroup

typedef __attribute__((ext_vector_type(16))) __bf16 v16bf;
typedef __attribute__((ext_vector_type(8)))  float  v8f;

union Frag { uint4 u[2]; v16bf v; };

__device__ __forceinline__ unsigned short f2bf(float f) {
  unsigned u = __float_as_uint(f);
  unsigned r = u + 0x7FFFu + ((u >> 16) & 1u);   // round-to-nearest-even
  return (unsigned short)(r >> 16);
}

__device__ __forceinline__ float sigmoid_(float x) {
  return 1.0f / (1.0f + __expf(-x));
}
__device__ __forceinline__ float tanh_(float x) {
  float e = __expf(2.0f * x);
  return (e - 1.0f) / (e + 1.0f);
}

// ------------------------- device-wide barrier -----------------------------
// bar[0] = arrival counter, bar[1] = generation. All blocks co-resident.
__device__ __forceinline__ void gbar(unsigned* bar, unsigned& lgen) {
  unsigned target = lgen + 1u;
  __threadfence();            // make this wave's stores visible device-wide
  __syncthreads();
  if (threadIdx.x == 0) {
    if (atomicAdd(&bar[0], 1u) == (unsigned)(NWG - 1)) {
      atomicExch(&bar[0], 0u);
      atomicAdd(&bar[1], 1u);
    } else {
      while (__hip_atomic_load(&bar[1], __ATOMIC_ACQUIRE,
                               __HIP_MEMORY_SCOPE_AGENT) < target) {
        __builtin_amdgcn_s_sleep(1);
      }
    }
  }
  lgen = target;
  __syncthreads();
}

// ------------------------- WMMA GEMM strip ---------------------------------
// Gout[0:256, nb:nb+16] = A(bf16 [256][512]) x W(LDS bf16 [16 n][512 k])^T
// Each of the 8 waves computes two 16x16 output tiles over K=512.
__device__ __forceinline__ void gemm_strip(const unsigned short* __restrict__ Ab,
                                           const unsigned short* __restrict__ Wl,
                                           float* __restrict__ Gout, int nb) {
  const int lane  = threadIdx.x & 31;
  const int wave  = threadIdx.x >> 5;
  const int nloc  = lane & 15;    // A: row-in-tile, B: n column, D: column
  const int khalf = lane >> 4;    // which K half-group this lane holds

  const int m0 = (wave * 2 + 0) * 16 + nloc;
  const int m1 = (wave * 2 + 1) * 16 + nloc;

  v8f acc0 = {}; v8f acc1 = {};

#pragma unroll
  for (int ks = 0; ks < HID / 32; ++ks) {
    const int k0 = ks * 32;

    // B fragment (32x16 KxN): lane holds K = khalf*8+[0..7] and 16+khalf*8+[0..7] for column nloc.
    Frag bfr;
    const uint4* bp = (const uint4*)(Wl + nloc * HID + k0 + khalf * 8);
    bfr.u[0] = bp[0];
    bfr.u[1] = bp[2];

    // A fragments (16x32 MxK): lane holds K = khalf*8+[0..7] and 16+khalf*8+[0..7].
    Frag a0, a1;
    const uint4* ap0 = (const uint4*)(Ab + m0 * HID + k0 + khalf * 8);
    a0.u[0] = ap0[0];
    a0.u[1] = ap0[2];      // +16 elements
    const uint4* ap1 = (const uint4*)(Ab + m1 * HID + k0 + khalf * 8);
    a1.u[0] = ap1[0];
    a1.u[1] = ap1[2];

    acc0 = __builtin_amdgcn_wmma_f32_16x16x32_bf16(false, a0.v, false, bfr.v,
                                                   (short)0, acc0, false, false);
    acc1 = __builtin_amdgcn_wmma_f32_16x16x32_bf16(false, a1.v, false, bfr.v,
                                                   (short)0, acc1, false, false);
  }

  // D layout: VGPR r, lanes 0-15 -> M=r, lanes 16-31 -> M=8+r; N = lane&15.
  const int col    = nb + nloc;
  const int rbase0 = (wave * 2 + 0) * 16 + khalf * 8;
  const int rbase1 = (wave * 2 + 1) * 16 + khalf * 8;
#pragma unroll
  for (int r = 0; r < 8; ++r) {
    Gout[(rbase0 + r) * G3 + col] = acc0[r];
    Gout[(rbase1 + r) * G3 + col] = acc1[r];
  }
}

// ------------------------- pointwise GRU updates ---------------------------
__device__ __forceinline__ void pointwise1(int t,
    const float* __restrict__ x,  const float* __restrict__ Wi1,
    const float* __restrict__ bi1, const float* __restrict__ bh1,
    const float* __restrict__ g1,
    float* __restrict__ h0f, unsigned short* __restrict__ h0b) {
  const int gid    = blockIdx.x * NTHREADS + threadIdx.x;
  const int stride = NWG * NTHREADS;
  for (int i = gid; i < BSZ * HID; i += stride) {
    const int b = i >> 9;
    const int j = i & (HID - 1);
    float hr = g1[b * G3 + j]            + bh1[j];
    float hz = g1[b * G3 + HID + j]      + bh1[HID + j];
    float hn = g1[b * G3 + 2 * HID + j]  + bh1[2 * HID + j];
    float xv = x[b * TSTEPS + t];
    float ir = xv * Wi1[j]           + bi1[j];
    float iz = xv * Wi1[HID + j]     + bi1[HID + j];
    float in_ = xv * Wi1[2 * HID + j] + bi1[2 * HID + j];
    float r = sigmoid_(ir + hr);
    float z = sigmoid_(iz + hz);
    float n = tanh_(in_ + r * hn);
    float h = (1.0f - z) * n + z * h0f[i];
    h0f[i] = h;
    h0b[i] = f2bf(h);
  }
}

__device__ __forceinline__ void pointwise2(
    const float* __restrict__ bi2, const float* __restrict__ bh2,
    const float* __restrict__ gi,  const float* __restrict__ gh,
    float* __restrict__ h1f, unsigned short* __restrict__ h1b) {
  const int gid    = blockIdx.x * NTHREADS + threadIdx.x;
  const int stride = NWG * NTHREADS;
  for (int i = gid; i < BSZ * HID; i += stride) {
    const int b = i >> 9;
    const int j = i & (HID - 1);
    float ir = gi[b * G3 + j]           + bi2[j];
    float iz = gi[b * G3 + HID + j]     + bi2[HID + j];
    float in_ = gi[b * G3 + 2 * HID + j] + bi2[2 * HID + j];
    float hr = gh[b * G3 + j]           + bh2[j];
    float hz = gh[b * G3 + HID + j]     + bh2[HID + j];
    float hn = gh[b * G3 + 2 * HID + j] + bh2[2 * HID + j];
    float r = sigmoid_(ir + hr);
    float z = sigmoid_(iz + hz);
    float n = tanh_(in_ + r * hn);
    float h = (1.0f - z) * n + z * h1f[i];
    h1f[i] = h;
    h1b[i] = f2bf(h);
  }
}

// ------------------------- init kernel -------------------------------------
__global__ void gru_init(const float* __restrict__ Wh1, const float* __restrict__ Wi2,
                         const float* __restrict__ Wh2,
                         const float* __restrict__ h0in, const float* __restrict__ h1in,
                         unsigned short* __restrict__ Wh1b, unsigned short* __restrict__ Wi2b,
                         unsigned short* __restrict__ Wh2b,
                         float* __restrict__ h0f, float* __restrict__ h1f,
                         unsigned short* __restrict__ h0b, unsigned short* __restrict__ h1b,
                         unsigned* __restrict__ bar) {
  const int gid    = blockIdx.x * blockDim.x + threadIdx.x;
  const int stride = gridDim.x * blockDim.x;
  for (int i = gid; i < G3 * HID; i += stride) {
    Wh1b[i] = f2bf(Wh1[i]);
    Wi2b[i] = f2bf(Wi2[i]);
    Wh2b[i] = f2bf(Wh2[i]);
  }
  for (int i = gid; i < BSZ * HID; i += stride) {
    float a = h0in[i], c = h1in[i];
    h0f[i] = a; h1f[i] = c;
    h0b[i] = f2bf(a); h1b[i] = f2bf(c);
  }
  if (gid == 0) { bar[0] = 0u; bar[1] = 0u; }
}

// ------------------------- persistent kernel -------------------------------
__global__ __launch_bounds__(NTHREADS, 1)
void gru_persistent(const float* __restrict__ x,
                    const float* __restrict__ Wi1, const float* __restrict__ bi1,
                    const float* __restrict__ bh1,
                    const float* __restrict__ bi2, const float* __restrict__ bh2,
                    const float* __restrict__ Wout, const float* __restrict__ bout,
                    const unsigned short* __restrict__ Wh1b,
                    const unsigned short* __restrict__ Wi2b,
                    const unsigned short* __restrict__ Wh2b,
                    float* __restrict__ h0f, float* __restrict__ h1f,
                    unsigned short* __restrict__ h0b, unsigned short* __restrict__ h1b,
                    float* __restrict__ g1, float* __restrict__ g2i,
                    float* __restrict__ g2h,
                    unsigned* __restrict__ bar,
                    float* __restrict__ out) {
  __shared__ unsigned short sWh1[NSTRIP * HID];   // 16 KB
  __shared__ unsigned short sWi2[NSTRIP * HID];   // 16 KB
  __shared__ unsigned short sWh2[NSTRIP * HID];   // 16 KB

  const int nb = blockIdx.x * NSTRIP;

  // Stage this WGP's weight strips into LDS once; resident for all steps.
  for (int i = threadIdx.x * 8; i < NSTRIP * HID; i += NTHREADS * 8) {
    *(uint4*)&sWh1[i] = *(const uint4*)&Wh1b[nb * HID + i];
    *(uint4*)&sWi2[i] = *(const uint4*)&Wi2b[nb * HID + i];
    *(uint4*)&sWh2[i] = *(const uint4*)&Wh2b[nb * HID + i];
  }
  __syncthreads();

  unsigned lgen = 0;
  for (int t = 0; t < TSTEPS; ++t) {
    // layer 1 recurrent GEMM: g1 = h0 @ Wh1^T (strip)
    gemm_strip(h0b, sWh1, g1, nb);
    gbar(bar, lgen);
    // layer 1 gate math -> h0
    pointwise1(t, x, Wi1, bi1, bh1, g1, h0f, h0b);
    gbar(bar, lgen);
    // layer 2 GEMMs: g2i = h0 @ Wi2^T, g2h = h1 @ Wh2^T (independent)
    gemm_strip(h0b, sWi2, g2i, nb);
    gemm_strip(h1b, sWh2, g2h, nb);
    gbar(bar, lgen);
    // layer 2 gate math -> h1
    pointwise2(bi2, bh2, g2i, g2h, h1f, h1b);
    gbar(bar, lgen);
  }

  // Final projection out = h1 @ W_out^T + b_out  (256 rows, one per thread)
  if (blockIdx.x == 0) {
    const int b = threadIdx.x;
    float acc = 0.0f;
    for (int k = 0; k < HID; k += 4) {
      acc += h1f[b * HID + k]     * Wout[k]
           + h1f[b * HID + k + 1] * Wout[k + 1]
           + h1f[b * HID + k + 2] * Wout[k + 2]
           + h1f[b * HID + k + 3] * Wout[k + 3];
    }
    out[b] = acc + bout[0];
  }
}

// ------------------------- host launcher -----------------------------------
extern "C" void kernel_launch(void* const* d_in, const int* in_sizes, int n_in,
                              void* d_out, int out_size, void* d_ws, size_t ws_size,
                              hipStream_t stream) {
  (void)in_sizes; (void)n_in; (void)out_size; (void)ws_size;

  const float* x    = (const float*)d_in[0];   // [256, 2048]
  const float* h0in = (const float*)d_in[1];   // [256, 512]
  const float* h1in = (const float*)d_in[2];   // [256, 512]
  const float* Wi1  = (const float*)d_in[3];   // [1536, 1]
  const float* Wh1  = (const float*)d_in[4];   // [1536, 512]
  const float* bi1  = (const float*)d_in[5];   // [1536]
  const float* bh1  = (const float*)d_in[6];   // [1536]
  const float* Wi2  = (const float*)d_in[7];   // [1536, 512]
  const float* Wh2  = (const float*)d_in[8];   // [1536, 512]
  const float* bi2  = (const float*)d_in[9];   // [1536]
  const float* bh2  = (const float*)d_in[10];  // [1536]
  const float* Wout = (const float*)d_in[11];  // [1, 512]
  const float* bout = (const float*)d_in[12];  // [1]
  float* out = (float*)d_out;                  // [256, 1]

  // Workspace layout (all offsets 256-aligned)
  char* ws = (char*)d_ws;
  size_t off = 0;
  auto take = [&](size_t bytes) { char* p = ws + off; off += (bytes + 255) & ~(size_t)255; return p; };
  unsigned*       bar  = (unsigned*)      take(256);
  float*          h0f  = (float*)         take((size_t)BSZ * HID * 4);
  float*          h1f  = (float*)         take((size_t)BSZ * HID * 4);
  unsigned short* h0b  = (unsigned short*)take((size_t)BSZ * HID * 2);
  unsigned short* h1b  = (unsigned short*)take((size_t)BSZ * HID * 2);
  unsigned short* Wh1b = (unsigned short*)take((size_t)G3 * HID * 2);
  unsigned short* Wi2b = (unsigned short*)take((size_t)G3 * HID * 2);
  unsigned short* Wh2b = (unsigned short*)take((size_t)G3 * HID * 2);
  float*          g1   = (float*)         take((size_t)BSZ * G3 * 4);
  float*          g2i  = (float*)         take((size_t)BSZ * G3 * 4);
  float*          g2h  = (float*)         take((size_t)BSZ * G3 * 4);

  gru_init<<<1024, 256, 0, stream>>>(Wh1, Wi2, Wh2, h0in, h1in,
                                     Wh1b, Wi2b, Wh2b,
                                     h0f, h1f, h0b, h1b, bar);

  gru_persistent<<<NWG, NTHREADS, 0, stream>>>(
      x, Wi1, bi1, bh1, bi2, bh2, Wout, bout,
      Wh1b, Wi2b, Wh2b,
      h0f, h1f, h0b, h1b,
      g1, g2i, g2h, bar, out);
}
